// CrossAttention_27358941675734
// MI455X (gfx1250) — matrix operand order repaired and checked by reading, not compiled, not run on hardware
//
#include <hip/hip_runtime.h>
#include <hip/hip_bf16.h>

#define E_   256
#define H_   8
#define D_   32
#define CH_  64
#define B_   4
#define Q_   900
#define K_   1024
#define QP_  912          // 57 tiles * 16
#define QTILES_ 57
#define KTILES_ 64
#define QGROUPS_ 15       // 15 groups * 4 waves * 16 rows = 960 >= 900
#define NKT_ (K_ / 32)    // 32 K-tiles of 32

typedef __attribute__((ext_vector_type(16))) _Float16 v16h;
typedef __attribute__((ext_vector_type(8)))  float    v8f;
typedef __attribute__((ext_vector_type(4)))  unsigned int u32x4;
typedef __attribute__((ext_vector_type(8)))  int          i32x8;
typedef __attribute__((ext_vector_type(4)))  int          i32x4;
typedef unsigned long long u64;

#if __has_builtin(__builtin_amdgcn_tensor_load_to_lds) && \
    __has_builtin(__builtin_amdgcn_s_wait_tensorcnt)
#define HAVE_TDM 1
#pragma message("CDNA5 probe: HAVE_TDM=1 (tensor_load_to_lds + s_wait_tensorcnt available)")
#else
#define HAVE_TDM 0
#pragma message("CDNA5 probe: HAVE_TDM=0 (cooperative-copy fallback in use)")
#endif
#if __clang_major__ >= 23
#pragma message("CDNA5 probe: clang >= 23 (6-arg TDM builtin form)")
#else
#pragma message("CDNA5 probe: clang < 23 (5-arg TDM builtin form)")
#endif

// ---------------------------------------------------------------------------
// Fragment layouts (CDNA5 wave32 WMMA, 16-bit, 16x16x32)
// A (16x32): lane L -> row m = L%16 ; halves K = (L/16)*16 + e
// B (32x16): lane L -> col n = L%16 ; halves K = (L/16)*16 + e
// C (16x16 f32): VGPR r, lane L -> (M = r + 8*(L/16), N = L%16)
// ---------------------------------------------------------------------------

__device__ __forceinline__ v16h load_a16_f32(const float* rowptr, int hi) {
  v16h a;
  const float* p0 = rowptr + hi * 16;
  const float* p1 = rowptr + hi * 16 + 8;
#pragma unroll
  for (int e = 0; e < 8; ++e) {
    a[e]     = (_Float16)p0[e];
    a[8 + e] = (_Float16)p1[e];
  }
  return a;
}

__device__ __forceinline__ v16h load_b16_f32(const float* wrow, int hi) {
  v16h b;
  const float* p = wrow + hi * 16;
#pragma unroll
  for (int e = 0; e < 16; ++e) b[e] = (_Float16)p[e];
  return b;
}

__device__ __forceinline__ v8f wmma16(v16h a, v16h b, v8f c) {
  return __builtin_amdgcn_wmma_f32_16x16x32_f16(false, a, false, b,
                                                (short)0, c, false, false);
}

#if HAVE_TDM
// Build a 2-D TDM descriptor (data_size = 2 bytes) and issue the DMA.
// group0: count=1 | lds_addr | global_addr[56:0] | type=2
// group1: data_size=1(2B) | tensor_dim0/1 | tile_dim0/1 | dim0 stride
__device__ __forceinline__ void tdm_load_2d(const void* gaddr, unsigned lds_addr,
                                            unsigned tensor_d0, unsigned tensor_d1,
                                            u64 stride0_elems,
                                            unsigned tile_d0, unsigned tile_d1) {
  const u64 ga = (u64)(uintptr_t)gaddr;
  u32x4 g0;
  g0[0] = 1u;                                              // count=1, user desc
  g0[1] = lds_addr;                                        // LDS byte address
  g0[2] = (unsigned)(ga & 0xffffffffu);                    // global_addr[31:0]
  g0[3] = (unsigned)((ga >> 32) & 0x01ffffffu) | (2u << 30); // addr[56:32]|type=2
  i32x8 g1;
  g1[0] = (int)(1u << 16);                                 // data_size=2B, mask=0
  g1[1] = (int)((tensor_d0 & 0xffffu) << 16);              // tensor_dim0[15:0]
  g1[2] = (int)((tensor_d0 >> 16) | ((tensor_d1 & 0xffffu) << 16));
  g1[3] = (int)((tensor_d1 >> 16) | (tile_d0 << 16));      // tile_dim0
  g1[4] = (int)(tile_d1 & 0xffffu);                        // tile_dim1, tile_dim2=0
  g1[5] = (int)(unsigned)(stride0_elems & 0xffffffffu);    // dim0 stride lo
  g1[6] = (int)(unsigned)((stride0_elems >> 32) & 0xffffu);
  g1[7] = 0;
  const i32x4 z4 = {0, 0, 0, 0};
#if __clang_major__ >= 23
  const i32x8 z8 = {0, 0, 0, 0, 0, 0, 0, 0};
  __builtin_amdgcn_tensor_load_to_lds(g0, g1, z4, z4, z8, 0);
#else
  __builtin_amdgcn_tensor_load_to_lds(g0, g1, z4, z4, 0);
#endif
}
#endif

// ---------------------------------------------------------------------------
// Kernel 1: q = nodes @ Wq^T + bq  ->  qws f16 [B,H,QP,D]  (pad rows zeroed)
// ---------------------------------------------------------------------------
__global__ void __launch_bounds__(256)
qproj_kernel(const float* __restrict__ nodes, const float* __restrict__ Wq,
             const float* __restrict__ bq, _Float16* __restrict__ qws) {
  const int b = blockIdx.x / QTILES_, tile = blockIdx.x % QTILES_;
  const int q0 = tile * 16;
  const int lane = threadIdx.x & 31, wv = threadIdx.x >> 5;  // wv == head
  const int hi = lane >> 4, n = lane & 15;
  const int j0 = wv * D_;

  const int rowq = q0 + n;
  const bool ok = rowq < Q_;
  const float* arow = nodes + ((size_t)b * Q_ + (ok ? rowq : 0)) * E_;

  v8f c0 = {}; v8f c1 = {};
  for (int kk = 0; kk < E_; kk += 32) {
    v16h a = load_a16_f32(arow + kk, hi);
    if (!ok) {
#pragma unroll
      for (int e = 0; e < 16; ++e) a[e] = (_Float16)0.f;
    }
    v16h b0 = load_b16_f32(Wq + (size_t)(j0 + n) * E_ + kk, hi);
    v16h b1 = load_b16_f32(Wq + (size_t)(j0 + 16 + n) * E_ + kk, hi);
    c0 = wmma16(a, b0, c0);
    c1 = wmma16(a, b1, c1);
  }
  const float bias0 = bq[j0 + n], bias1 = bq[j0 + 16 + n];
#pragma unroll
  for (int r = 0; r < 8; ++r) {
    const int M = r + 8 * hi;
    const int qg = q0 + M;
    const bool rok = qg < Q_;
    const size_t base = (((size_t)b * H_ + wv) * QP_ + qg) * D_;
    qws[base + n]      = (_Float16)(rok ? c0[r] + bias0 : 0.f);
    qws[base + 16 + n] = (_Float16)(rok ? c1[r] + bias1 : 0.f);
  }
}

// ---------------------------------------------------------------------------
// Kernel 2: kv = images @ Wkv^T + bkv
//   k -> kws f16 [B,H,K,D] ; v -> vT f16 [B,H,D,K] (transposed)
// ---------------------------------------------------------------------------
__global__ void __launch_bounds__(256)
kvproj_kernel(const float* __restrict__ images, const float* __restrict__ Wkv,
              const float* __restrict__ bkv, _Float16* __restrict__ kws,
              _Float16* __restrict__ vT) {
  const int b = blockIdx.x / KTILES_, tile = blockIdx.x % KTILES_;
  const int k0 = tile * 16;
  const int lane = threadIdx.x & 31, wv = threadIdx.x >> 5;
  const int hi = lane >> 4, n = lane & 15;
  const int j0 = wv * D_;

  const float* arow = images + ((size_t)b * K_ + k0 + n) * E_;

  v8f ck0 = {}, ck1 = {}, cv0 = {}, cv1 = {};
  for (int kk = 0; kk < E_; kk += 32) {
    v16h a  = load_a16_f32(arow + kk, hi);
    v16h b0 = load_b16_f32(Wkv + (size_t)(j0 + n) * E_ + kk, hi);
    v16h b1 = load_b16_f32(Wkv + (size_t)(j0 + 16 + n) * E_ + kk, hi);
    v16h b2 = load_b16_f32(Wkv + (size_t)(E_ + j0 + n) * E_ + kk, hi);
    v16h b3 = load_b16_f32(Wkv + (size_t)(E_ + j0 + 16 + n) * E_ + kk, hi);
    ck0 = wmma16(a, b0, ck0);
    ck1 = wmma16(a, b1, ck1);
    cv0 = wmma16(a, b2, cv0);
    cv1 = wmma16(a, b3, cv1);
  }
  const float bk0 = bkv[j0 + n], bk1 = bkv[j0 + 16 + n];
  const float bv0 = bkv[E_ + j0 + n], bv1 = bkv[E_ + j0 + 16 + n];
#pragma unroll
  for (int r = 0; r < 8; ++r) {
    const int M = r + 8 * hi;
    const int krow = k0 + M;
    const size_t kbase = (((size_t)b * H_ + wv) * K_ + krow) * D_;
    kws[kbase + n]      = (_Float16)(ck0[r] + bk0);
    kws[kbase + 16 + n] = (_Float16)(ck1[r] + bk1);
    const size_t vb = ((size_t)b * H_ + wv) * D_;
    vT[(vb + n) * K_ + krow]      = (_Float16)(cv0[r] + bv0);
    vT[(vb + 16 + n) * K_ + krow] = (_Float16)(cv1[r] + bv1);
  }
}

// ---------------------------------------------------------------------------
// Kernel 3: fused flash attention + CPB MLP, TDM double-buffered K/V staging
// ---------------------------------------------------------------------------
__global__ void __launch_bounds__(128)
attn_kernel(const _Float16* __restrict__ qws, const _Float16* __restrict__ kws,
            const _Float16* __restrict__ vT, const float* __restrict__ rel,
            const unsigned char* __restrict__ mask,
            const float* __restrict__ Wc1, const float* __restrict__ bc1,
            const float* __restrict__ Wc2, const float* __restrict__ bc2,
            float* __restrict__ aout) {
  __shared__ float4 tab[CH_];                         // (w1x, w1y, b1, w2[h])
  __shared__ _Float16 pbuf[4][16 * D_];               // per-wave prob tile
  __shared__ alignas(32) _Float16 kstage[2][32 * D_]; // [klocal][d]
  __shared__ alignas(32) _Float16 vstage[2][32 * D_]; // [d][klocal]

  const int g  = blockIdx.x % QGROUPS_;
  const int bh = blockIdx.x / QGROUPS_;
  const int b = bh / H_, h = bh % H_;
  const int tid = threadIdx.x;
  const int lane = tid & 31, wv = tid >> 5;
  const int hi = lane >> 4, n = lane & 15;

  if (tid < CH_) {
    tab[tid] = make_float4(Wc1[tid * 2], Wc1[tid * 2 + 1], bc1[tid],
                           Wc2[h * CH_ + tid]);
  }

  int q0 = g * 64 + wv * 16;
  if (q0 >= Q_) q0 = QP_ - 16;   // duplicate last tile; stores value-identical

  const float scale = 0.17677669529663687f;  // 1/sqrt(32)
  const float bc2h = bc2[h];
  const float NEG_INF = -__builtin_inff();
  const size_t bhq = (size_t)b * H_ + h;

  const _Float16* kbase = kws + bhq * (size_t)K_ * D_;
  const _Float16* vbase = vT + bhq * (size_t)D_ * K_;
  const float* relb = rel + (size_t)b * Q_ * K_ * 2;
  const unsigned char* mb = mask + (size_t)b * K_;

  // q A-fragment (row m = n, K-dim = head_dim)
  const _Float16* qrow = qws + (bhq * QP_ + q0 + n) * D_;
  v16h qa;
#pragma unroll
  for (int e = 0; e < 8; ++e) {
    qa[e]     = qrow[hi * 16 + e];
    qa[8 + e] = qrow[hi * 16 + 8 + e];
  }

  // ---- K/V tile staging: TDM DMA (preferred) or cooperative copy ----
  auto issue_stage = [&](int ktn, int buf) {
#if HAVE_TDM
    if (wv == 0) {
      // K tile: tensor [K,D] contiguous rows; tile 32x32, dim0 stride = D
      tdm_load_2d(kbase + (size_t)ktn * 32 * D_,
                  (unsigned)(uintptr_t)&kstage[buf][0],
                  D_, K_, (u64)D_, D_, 32);
      // V tile: tensor [D,K]; tile 32 cols x 32 rows, dim0 stride = K
      tdm_load_2d(vbase + (size_t)ktn * 32,
                  (unsigned)(uintptr_t)&vstage[buf][0],
                  K_, D_, (u64)K_, 32, 32);
    }
#else
    {  // cooperative fallback: 128 threads x (2 + 2) u64 chunks
      const u64* sk = (const u64*)(kbase + (size_t)ktn * 32 * D_);
      u64* dk = (u64*)&kstage[buf][0];
      u64* dv = (u64*)&vstage[buf][0];
#pragma unroll
      for (int i = tid; i < 256; i += 128) dk[i] = sk[i];
#pragma unroll
      for (int i = tid; i < 256; i += 128) {
        const int row = i >> 3, ch = i & 7;
        dv[i] = ((const u64*)(vbase + (size_t)row * K_ + ktn * 32))[ch];
      }
    }
#endif
  };

  issue_stage(0, 0);
#if HAVE_TDM
  if (wv == 0) __builtin_amdgcn_s_wait_tensorcnt(0);
#endif
  __syncthreads();   // also publishes tab[]

  float mrow[8], lrow[8];
  v8f o0 = {}, o1 = {};
#pragma unroll
  for (int r = 0; r < 8; ++r) { mrow[r] = NEG_INF; lrow[r] = 0.f; }

  for (int kt = 0; kt < NKT_; ++kt) {
    const int cur = kt & 1, nxt = cur ^ 1;
    if (kt + 1 < NKT_) issue_stage(kt + 1, nxt);  // overlap DMA with compute

    const int kg0 = kt * 32 + n;
    const int kg1 = kg0 + 16;

    // ---- scores: K/V fragments now come from LDS ----
    v16h kb0 = *(const v16h*)(&kstage[cur][n * D_ + hi * 16]);
    v16h kb1 = *(const v16h*)(&kstage[cur][(16 + n) * D_ + hi * 16]);
    v8f s0 = {}, s1 = {};
    s0 = wmma16(qa, kb0, s0);
    s1 = wmma16(qa, kb1, s1);

    // ---- CPB MLP, 16 elements/lane, one LDS table entry per j ----
    float x0[8], y0[8], x1[8], y1[8];
#pragma unroll
    for (int r = 0; r < 8; ++r) {
      const int qg = q0 + 8 * hi + r;
      if (qg < Q_) {
        const float* rp0 = relb + ((size_t)qg * K_ + kg0) * 2;
        const float* rp1 = relb + ((size_t)qg * K_ + kg1) * 2;
        x0[r] = rp0[0]; y0[r] = rp0[1];
        x1[r] = rp1[0]; y1[r] = rp1[1];
        if (r == 0 && kt + 2 < NKT_)  // prefetch the dominant HBM stream
          __builtin_prefetch(rp0 + 128, 0, 1);
      } else {
        x0[r] = y0[r] = x1[r] = y1[r] = 0.f;
      }
    }
    float a0[8] = {}, a1[8] = {};
#pragma unroll 4
    for (int j = 0; j < CH_; ++j) {
      const float4 t = tab[j];
#pragma unroll
      for (int r = 0; r < 8; ++r) {
        float h0 = fmaf(t.x, x0[r], fmaf(t.y, y0[r], t.z));
        h0 = fmaxf(h0, 0.f);
        a0[r] = fmaf(t.w, h0, a0[r]);
        float h1 = fmaf(t.x, x1[r], fmaf(t.y, y1[r], t.z));
        h1 = fmaxf(h1, 0.f);
        a1[r] = fmaf(t.w, h1, a1[r]);
      }
    }
    const bool msk0 = mb[kg0] != 0, msk1 = mb[kg1] != 0;
#pragma unroll
    for (int r = 0; r < 8; ++r) {
      const float v0 = fmaf(s0[r], scale, a0[r] + bc2h);
      const float v1 = fmaf(s1[r], scale, a1[r] + bc2h);
      s0[r] = msk0 ? NEG_INF : v0;
      s1[r] = msk1 ? NEG_INF : v1;
    }

    // ---- online softmax: row reductions across the 16-lane half ----
    float p0[8], p1[8];
#pragma unroll
    for (int r = 0; r < 8; ++r) {
      float sm = fmaxf(s0[r], s1[r]);
#pragma unroll
      for (int msh = 1; msh < 16; msh <<= 1)
        sm = fmaxf(sm, __shfl_xor(sm, msh, 32));
      const float mnew = fmaxf(mrow[r], sm);
      float alpha, e0, e1;
      if (mnew == NEG_INF) {
        alpha = 1.f; e0 = 0.f; e1 = 0.f;
      } else {
        alpha = __expf(mrow[r] - mnew);
        e0 = __expf(s0[r] - mnew);
        e1 = __expf(s1[r] - mnew);
      }
      float ls = e0 + e1;
#pragma unroll
      for (int msh = 1; msh < 16; msh <<= 1)
        ls += __shfl_xor(ls, msh, 32);
      mrow[r] = mnew;
      lrow[r] = lrow[r] * alpha + ls;
      o0[r] = o0[r] * alpha;
      o1[r] = o1[r] * alpha;
      p0[r] = e0; p1[r] = e1;
    }

    // ---- re-stripe probs (C layout -> A layout) via per-wave LDS ----
    _Float16* pb = &pbuf[wv][0];
#pragma unroll
    for (int r = 0; r < 8; ++r) {
      const int M = 8 * hi + r;
      pb[M * D_ + n]      = (_Float16)p0[r];
      pb[M * D_ + 16 + n] = (_Float16)p1[r];
    }
    v16h pa;
    const _Float16* pr = pb + n * D_;
#pragma unroll
    for (int e = 0; e < 8; ++e) {
      pa[e]     = pr[hi * 16 + e];
      pa[8 + e] = pr[hi * 16 + 8 + e];
    }

    // ---- attn @ V from staged transposed tile ----
    v16h vb0 = *(const v16h*)(&vstage[cur][n * D_ + hi * 16]);
    v16h vb1 = *(const v16h*)(&vstage[cur][(16 + n) * D_ + hi * 16]);
    o0 = wmma16(pa, vb0, o0);
    o1 = wmma16(pa, vb1, o1);

    // publish next buffer / protect current one before it is overwritten
#if HAVE_TDM
    if (wv == 0) __builtin_amdgcn_s_wait_tensorcnt(0);
#endif
    __syncthreads();
  }

  // ---- epilogue: normalize and scatter to [B,Q,E] staging ----
#pragma unroll
  for (int r = 0; r < 8; ++r) {
    const int M = 8 * hi + r;
    const int qg = q0 + M;
    if (qg < Q_) {
      const float inv = lrow[r] > 0.f ? 1.f / lrow[r] : 0.f;
      const size_t ob = ((size_t)b * Q_ + qg) * E_ + h * D_;
      aout[ob + n]      = o0[r] * inv;
      aout[ob + 16 + n] = o1[r] * inv;
    }
  }
}

// ---------------------------------------------------------------------------
// Kernel 4: out = aout @ Wo^T + bo
// ---------------------------------------------------------------------------
__global__ void __launch_bounds__(256)
outproj_kernel(const float* __restrict__ aout, const float* __restrict__ Wo,
               const float* __restrict__ bo, float* __restrict__ out) {
  const int b = blockIdx.x / QTILES_, tile = blockIdx.x % QTILES_;
  const int q0 = tile * 16;
  const int lane = threadIdx.x & 31, wv = threadIdx.x >> 5;
  const int hi = lane >> 4, n = lane & 15;
  const int j0 = wv * D_;

  const int rowq = q0 + n;
  const bool ok = rowq < Q_;
  const float* arow = aout + ((size_t)b * Q_ + (ok ? rowq : 0)) * E_;

  v8f c0 = {}, c1 = {};
  for (int kk = 0; kk < E_; kk += 32) {
    v16h a = load_a16_f32(arow + kk, hi);
    if (!ok) {
#pragma unroll
      for (int e = 0; e < 16; ++e) a[e] = (_Float16)0.f;
    }
    v16h b0 = load_b16_f32(Wo + (size_t)(j0 + n) * E_ + kk, hi);
    v16h b1 = load_b16_f32(Wo + (size_t)(j0 + 16 + n) * E_ + kk, hi);
    c0 = wmma16(a, b0, c0);
    c1 = wmma16(a, b1, c1);
  }
  const float bb0 = bo[j0 + n], bb1 = bo[j0 + 16 + n];
#pragma unroll
  for (int r = 0; r < 8; ++r) {
    const int M = r + 8 * hi;
    const int qg = q0 + M;
    if (qg < Q_) {
      const size_t base = ((size_t)b * Q_ + qg) * E_;
      out[base + j0 + n]      = c0[r] + bb0;
      out[base + j0 + 16 + n] = c1[r] + bb1;
    }
  }
}

// ---------------------------------------------------------------------------
extern "C" void kernel_launch(void* const* d_in, const int* in_sizes, int n_in,
                              void* d_out, int out_size, void* d_ws,
                              size_t ws_size, hipStream_t stream) {
  (void)in_sizes; (void)n_in; (void)out_size; (void)ws_size;
  const float* nodes  = (const float*)d_in[0];
  const float* images = (const float*)d_in[1];
  const unsigned char* msk = (const unsigned char*)d_in[2];
  const float* rel = (const float*)d_in[3];
  const float* Wq  = (const float*)d_in[4];
  const float* bq  = (const float*)d_in[5];
  const float* Wkv = (const float*)d_in[6];
  const float* bkv = (const float*)d_in[7];
  const float* Wc1 = (const float*)d_in[8];
  const float* bc1 = (const float*)d_in[9];
  const float* Wc2 = (const float*)d_in[10];
  const float* bc2 = (const float*)d_in[11];
  const float* Wo  = (const float*)d_in[12];
  const float* bo  = (const float*)d_in[13];

  char* ws = (char*)d_ws;
  size_t off = 0;
  _Float16* qws = (_Float16*)(ws + off); off += (size_t)B_ * H_ * QP_ * D_ * 2;
  _Float16* kws = (_Float16*)(ws + off); off += (size_t)B_ * H_ * K_ * D_ * 2;
  _Float16* vT  = (_Float16*)(ws + off); off += (size_t)B_ * H_ * D_ * K_ * 2;
  float* aout   = (float*)(ws + off);    off += (size_t)B_ * Q_ * E_ * 4;

  qproj_kernel<<<B_ * QTILES_, 256, 0, stream>>>(nodes, Wq, bq, qws);
  kvproj_kernel<<<B_ * KTILES_, 256, 0, stream>>>(images, Wkv, bkv, kws, vT);
  attn_kernel<<<B_ * H_ * QGROUPS_, 128, 0, stream>>>(qws, kws, vT, rel, msk,
                                                      Wc1, bc1, Wc2, bc2, aout);
  outproj_kernel<<<B_ * QTILES_, 256, 0, stream>>>(aout, Wo, bo, (float*)d_out);
}
